// LSHAttention_33208687132794
// MI455X (gfx1250) — matrix operand order repaired and checked by reading, not compiled, not run on hardware
//
#include <hip/hip_runtime.h>
#include <hip/hip_bf16.h>

#define B_    8
#define L_    4096
#define D_    64
#define NH_   8
#define NB_R  32                 // n_buckets/2 rotations per hash
#define TOT_  (NH_*L_)           // 32768 sorted slots per batch
#define CH_   128                // queries per chunk (2*BUCKET)
#define KC_   256                // keys per chunk (look-one-back)
#define NCHB_ (TOT_/CH_)         // 256 chunks per batch
#define SCALE_ 0.125f            // D^-0.5
#define NEGB_  (-1e9f)

typedef __attribute__((ext_vector_type(8)))  float  v8f;
typedef __attribute__((ext_vector_type(16))) __bf16 v16bf;
typedef __attribute__((ext_vector_type(8)))  __bf16 v8bf;

__device__ __forceinline__ v16bf cat8(v8bf lo, v8bf hi) {
  return __builtin_shufflevector(lo, hi, 0,1,2,3,4,5,6,7,8,9,10,11,12,13,14,15);
}

// ---------------------------------------------------------------------------
// Kernel 1: normalize qk rows, compute LSH bucket per (b, h, l).
// One wave32 per (b,l). Lane r computes rotation r for each hash round.
// ---------------------------------------------------------------------------
__global__ void __launch_bounds__(256) k_hash(
    const float* __restrict__ qk, const float* __restrict__ rot,
    float* __restrict__ qkn, int* __restrict__ bkt)
{
  __shared__ float row[8][D_];
  int w = threadIdx.x >> 5, lane = threadIdx.x & 31;
  int gid = blockIdx.x * 8 + w;
  int b = gid / L_, l = gid % L_;

  const float2 q2 = *(const float2*)(qk + ((size_t)b*L_ + l)*D_ + lane*2);
  float ss = q2.x*q2.x + q2.y*q2.y;
  #pragma unroll
  for (int m=16; m; m>>=1) ss += __shfl_xor(ss, m, 32);
  float inv = 1.0f / fmaxf(sqrtf(ss), 1e-12f);
  float2 qn = make_float2(q2.x*inv, q2.y*inv);
  *(float2*)(qkn + ((size_t)b*L_ + l)*D_ + lane*2) = qn;
  row[w][lane*2] = qn.x; row[w][lane*2+1] = qn.y;   // wave-private, DS in-order

  #pragma unroll 1
  for (int h=0; h<NH_; ++h) {
    float x = 0.f;
    #pragma unroll
    for (int d=0; d<D_; ++d)
      x += row[w][d] * rot[d*(NH_*NB_R) + h*NB_R + lane];
    // argmax over concat([x_r], [-x_r]); first occurrence wins ties
    float val = x; int idx = lane;
    if (-x > x) { val = -x; idx = lane + NB_R; }
    #pragma unroll
    for (int m=16; m; m>>=1) {
      float v2 = __shfl_xor(val, m, 32);
      int   i2 = __shfl_xor(idx, m, 32);
      if (v2 > val || (v2 == val && i2 < idx)) { val = v2; idx = i2; }
    }
    if (lane == 0) bkt[((size_t)b*NH_ + h)*L_ + l] = idx;
  }
}

// ---------------------------------------------------------------------------
// Kernel 2: stable counting sort per batch. Order = (bucket, l, h) ascending,
// identical to stable argsort of key L*bucket + l over the (h,l)-flattened
// array. One block per batch; 128 l-blocks of 32 l's (x 8 h) each owned by
// one thread, so no atomics and fully deterministic.
// ---------------------------------------------------------------------------
__global__ void __launch_bounds__(256) k_sort(
    const int* __restrict__ bkt, int* __restrict__ sidx)
{
  __shared__ int hist[64*128];
  __shared__ int cnt[64];
  __shared__ int start[64];
  int b = blockIdx.x, t = threadIdx.x;

  for (int i=t; i<64*128; i+=256) hist[i] = 0;
  __syncthreads();

  if (t < 128) {
    for (int li=0; li<32; ++li) {
      int l = t*32 + li;
      for (int h=0; h<NH_; ++h)
        hist[bkt[((size_t)b*NH_ + h)*L_ + l]*128 + t]++;
    }
  }
  __syncthreads();

  if (t < 64) {
    int s = 0;
    for (int blk=0; blk<128; ++blk) s += hist[t*128 + blk];
    cnt[t] = s;
  }
  __syncthreads();
  if (t == 0) {
    int run = 0;
    for (int k=0; k<64; ++k) { start[k] = run; run += cnt[k]; }
  }
  __syncthreads();
  if (t < 64) {
    int run = start[t];
    for (int blk=0; blk<128; ++blk) {
      int tmp = hist[t*128 + blk]; hist[t*128 + blk] = run; run += tmp;
    }
  }
  __syncthreads();

  if (t < 128) {
    for (int li=0; li<32; ++li) {          // scan in (l, h) order => stable
      int l = t*32 + li;
      for (int h=0; h<NH_; ++h) {
        int bk = bkt[((size_t)b*NH_ + h)*L_ + l];
        int pos = hist[bk*128 + t]++;
        sidx[(size_t)b*TOT_ + pos] = h*L_ + l;
      }
    }
  }
}

// ---------------------------------------------------------------------------
// Kernel 3: chunked attention. One block (8 waves) per (b, chunk).
// Wave w owns 16 query rows. bf16 WMMA for QK^T and PV, fp32 softmax.
// ---------------------------------------------------------------------------
__global__ void __launch_bounds__(256) k_attn(
    const float* __restrict__ qkn, const float* __restrict__ v,
    const unsigned char* __restrict__ mask, const int* __restrict__ sidx,
    float* __restrict__ outUns)
{
  __shared__ __align__(32) __bf16 sK[KC_*D_];   // keys [256][64]; reused as VT[64][256]
  __shared__ float sBias[KC_];
  __shared__ int   sI[KC_];
  __shared__ __align__(32) __bf16 sP[8*16*32];  // per-wave P relayout slice

  int b = blockIdx.x / NCHB_;
  int c = blockIdx.x % NCHB_;
  int t = threadIdx.x;

  // ---- Phase A: gather keys = [prev chunk ; current chunk] into LDS (bf16)
  {
    int rowp = (t < CH_) ? (((c + NCHB_ - 1) % NCHB_)*CH_ + t)
                         : (c*CH_ + (t - CH_));
    int idx = sidx[(size_t)b*TOT_ + rowp];
    int l = idx % L_;
    sI[t]   = idx;
    sBias[t] = mask[(size_t)b*L_ + l] ? 0.f : NEGB_;
    const float4* src = (const float4*)(qkn + ((size_t)b*L_ + l)*D_);
    __builtin_prefetch(v + ((size_t)b*L_ + l)*D_, 0, 1);   // global_prefetch_b8
    #pragma unroll
    for (int i=0; i<16; ++i) {
      float4 f = src[i]; int d = i*4;
      sK[t*D_+d+0]=(__bf16)f.x; sK[t*D_+d+1]=(__bf16)f.y;
      sK[t*D_+d+2]=(__bf16)f.z; sK[t*D_+d+3]=(__bf16)f.w;
    }
  }
  __syncthreads();

  int w = t >> 5, lane = t & 31;
  int hi = (lane >= 16) ? 1 : 0;
  int ln = lane & 15;
  int qBase = w * 16;

  // ---- Phase B: S = (scale*Q) K^T, 16 key-tiles x 2 WMMAs (d split 0/32)
  const __bf16* qrow = sK + (CH_ + qBase + ln)*D_;  // queries = keys[128..255]
  v16bf aq0 = *(const v16bf*)(qrow +      (hi?16:0));
  v16bf aq1 = *(const v16bf*)(qrow + 32 + (hi?16:0));

  v8f acc[16];
  #pragma unroll
  for (int kt=0; kt<16; ++kt) {
    const __bf16* krow = sK + (kt*16 + ln)*D_;
    v16bf b0 = *(const v16bf*)(krow +      (hi?16:0));
    v16bf b1 = *(const v16bf*)(krow + 32 + (hi?16:0));
    v8f z = {};
    acc[kt] = __builtin_amdgcn_wmma_f32_16x16x32_bf16(false, aq0, false, b0,
                                                      (short)0, z, false, false);
    acc[kt] = __builtin_amdgcn_wmma_f32_16x16x32_bf16(false, aq1, false, b1,
                                                      (short)0, acc[kt], false, false);
  }

  // ---- softmax over 256 keys; row (j + 8*hi) lives across a 16-lane half
  float rmax[8], rsum[8], inv_[8];
  #pragma unroll
  for (int j=0; j<8; ++j) rmax[j] = -3.4e38f;
  #pragma unroll
  for (int kt=0; kt<16; ++kt) {
    float bias = sBias[kt*16 + ln];
    #pragma unroll
    for (int j=0; j<8; ++j) {
      float s = acc[kt][j]*SCALE_ + bias;
      acc[kt][j] = s;
      rmax[j] = fmaxf(rmax[j], s);
    }
  }
  #pragma unroll
  for (int j=0; j<8; ++j) {
    float m = rmax[j];
    #pragma unroll
    for (int x=8; x; x>>=1) m = fmaxf(m, __shfl_xor(m, x, 16));
    rmax[j] = m; rsum[j] = 0.f;
  }
  #pragma unroll
  for (int kt=0; kt<16; ++kt)
    #pragma unroll
    for (int j=0; j<8; ++j) {
      float e = __expf(acc[kt][j] - rmax[j]);
      acc[kt][j] = e; rsum[j] += e;
    }
  #pragma unroll
  for (int j=0; j<8; ++j) {
    float s = rsum[j];
    #pragma unroll
    for (int x=8; x; x>>=1) s += __shfl_xor(s, x, 16);
    inv_[j] = 1.0f / s;
  }
  __syncthreads();   // all waves done with sK (keys)

  // ---- Phase C: gather V transposed into the same LDS: VT[d][key]
  {
    int l = sI[t] % L_;
    const float4* src = (const float4*)(v + ((size_t)b*L_ + l)*D_);
    #pragma unroll
    for (int i=0; i<16; ++i) {
      float4 f = src[i]; int d = i*4;
      sK[(d+0)*KC_ + t] = (__bf16)f.x;
      sK[(d+1)*KC_ + t] = (__bf16)f.y;
      sK[(d+2)*KC_ + t] = (__bf16)f.z;
      sK[(d+3)*KC_ + t] = (__bf16)f.w;
    }
  }
  __syncthreads();

  // ---- Phase D: O = P V, relayout P via wave-private LDS slice per k-step
  v8f oacc[4];
  #pragma unroll
  for (int dt=0; dt<4; ++dt) { v8f z = {}; oacc[dt] = z; }
  __bf16* pw = sP + w*(16*32);
  #pragma unroll
  for (int step=0; step<8; ++step) {
    int kOff = step*32;
    #pragma unroll
    for (int j=0; j<8; ++j) {              // C-layout -> row-major bf16 slice
      int r = j + (hi ? 8 : 0);
      pw[r*32 + ln]      = (__bf16)(acc[2*step  ][j] * inv_[j]);
      pw[r*32 + 16 + ln] = (__bf16)(acc[2*step+1][j] * inv_[j]);
    }
    const __bf16* prow = pw + ln*32;       // A-layout reload (same wave, DS in-order)
    v16bf ap = *(const v16bf*)(prow + (hi?16:0));
    #pragma unroll
    for (int dt=0; dt<4; ++dt) {
      const __bf16* vtrow = sK + (dt*16 + ln)*KC_ + kOff + (hi?16:0);
      v16bf bv = *(const v16bf*)vtrow;
      oacc[dt] = __builtin_amdgcn_wmma_f32_16x16x32_bf16(false, ap, false, bv,
                                                         (short)0, oacc[dt], false, false);
    }
  }

  // ---- Phase E: scatter rows to unsorted (b, h*L+l, d) workspace
  #pragma unroll
  for (int j=0; j<8; ++j) {
    int m = j + (hi ? 8 : 0);
    int idx = sI[CH_ + qBase + m];
    float* dst = outUns + ((size_t)b*TOT_ + idx)*D_ + ln;
    dst[0]  = oacc[0][j];
    dst[16] = oacc[1][j];
    dst[32] = oacc[2][j];
    dst[48] = oacc[3][j];
  }
}

// ---------------------------------------------------------------------------
// Kernel 4: average the 8 hash rounds.
// ---------------------------------------------------------------------------
__global__ void __launch_bounds__(256) k_reduce(
    const float* __restrict__ outUns, float* __restrict__ out)
{
  size_t i = (size_t)blockIdx.x*256 + threadIdx.x;   // over B*L*D
  int d = (int)(i % D_);
  size_t bl = i / D_;
  int l = (int)(bl % L_);
  int b = (int)(bl / L_);
  float s = 0.f;
  #pragma unroll
  for (int h=0; h<NH_; ++h)
    s += outUns[((size_t)b*TOT_ + (size_t)h*L_ + l)*D_ + d];
  out[i] = s * (1.0f/NH_);
}

extern "C" void kernel_launch(void* const* d_in, const int* in_sizes, int n_in,
                              void* d_out, int out_size, void* d_ws, size_t ws_size,
                              hipStream_t stream)
{
  (void)in_sizes; (void)n_in; (void)out_size; (void)ws_size;
  const float* qk  = (const float*)d_in[0];
  const float* v   = (const float*)d_in[1];
  const unsigned char* mask = (const unsigned char*)d_in[2];
  const float* rot = (const float*)d_in[3];

  float* qkn    = (float*)d_ws;                       // B*L*D f32  (8 MB)
  int*   bkt    = (int*)(qkn + (size_t)B_*L_*D_);     // B*NH*L i32 (1 MB)
  int*   sidx   = bkt + (size_t)B_*NH_*L_;            // B*TOT i32  (1 MB)
  float* outUns = (float*)(sidx + (size_t)B_*TOT_);   // B*TOT*D f32 (64 MB)
  float* out    = (float*)d_out;

  k_hash  <<<(B_*L_)/8,    256, 0, stream>>>(qk, rot, qkn, bkt);
  k_sort  <<<B_,           256, 0, stream>>>(bkt, sidx);
  k_attn  <<<B_*NCHB_,     256, 0, stream>>>(qkn, v, mask, sidx, outUns);
  k_reduce<<<(B_*L_*D_)/256, 256, 0, stream>>>(outUns, out);
}
